// MoeLayer_23373212025093
// MI455X (gfx1250) — matrix operand order repaired and checked by reading, not compiled, not run on hardware
//
#include <hip/hip_runtime.h>
#include <hip/hip_bf16.h>

typedef __attribute__((ext_vector_type(16))) __bf16 v16bf;
typedef __attribute__((ext_vector_type(8)))  float  v8f;

#define N_EXPERTS 8
#define D_MODEL   1024
#define D_FF      4096
#define T_TOKENS  2048
#define SLOTS     (2 * T_TOKENS)   // 4096 routed (token,expert) slots
#define KST       40               // LDS row stride in halves (80B: 16B-aligned, conflict-free)

// ---------- helpers ----------

__device__ inline unsigned short f2bf(float f) {
  unsigned int u = __float_as_uint(f);
  unsigned int r = u + 0x7FFFu + ((u >> 16) & 1u);   // round-to-nearest-even
  return (unsigned short)(r >> 16);
}

__device__ inline v8f zero_v8f() {
  v8f z = {0.f, 0.f, 0.f, 0.f, 0.f, 0.f, 0.f, 0.f};
  return z;
}

// CDNA5 async global->LDS copy, 16B per lane, tracked by ASYNCcnt.
__device__ __forceinline__ void async_copy16(const unsigned short* lds_dst,
                                             const unsigned short* gsrc) {
  unsigned ldsa = (unsigned)(unsigned long long)lds_dst;  // low 32 bits = LDS byte addr
  asm volatile("global_load_async_to_lds_b128 %0, %1, off"
               :: "v"(ldsa), "v"(gsrc)
               : "memory");
}

#define WAIT_ASYNC(n) asm volatile("s_wait_asynccnt " #n ::: "memory")

// A fragment: 16x32 bf16, lane = M (row), per ISA: VGPR v (half p) holds
// K = (v&3)*2 + p + (v>=4 ? 16 : 0) + (lane>=16 ? 8 : 0)
__device__ __forceinline__ v16bf load_frag_a(const unsigned short* p, int lane) {
  int row  = lane & 15;
  int half = lane >> 4;
  union { v16bf v; unsigned int w[8]; } f;
  const unsigned short* rp = p + row * KST + half * 8;
#pragma unroll
  for (int v = 0; v < 8; ++v) {
    int ko = ((v & 3) * 2) + ((v >> 2) * 16);
    f.w[v] = *reinterpret_cast<const unsigned int*>(rp + ko);
  }
  return f.v;
}

// B fragment: 32x16 bf16 stored K-contiguous per column in LDS ([n][k]).
// lane = N (col); VGPR v (half p): k = (v&3)*2 + p + (v>=4 ? 16 : 0) + (lane>=16 ? 8 : 0)
__device__ __forceinline__ v16bf load_frag_b(const unsigned short* p, int lane) {
  int col  = lane & 15;
  int half = lane >> 4;
  union { v16bf v; unsigned int w[8]; } f;
  const unsigned short* rp = p + col * KST + half * 8;
#pragma unroll
  for (int v = 0; v < 8; ++v) {
    f.w[v] = *reinterpret_cast<const unsigned int*>(rp + ((v & 3) * 2) + ((v >> 2) * 16));
  }
  return f.v;
}

// ---------- small kernels ----------

__global__ void moe_init_kernel(float* __restrict__ out, int n, int* __restrict__ cnt) {
  int i = blockIdx.x * blockDim.x + threadIdx.x;
  if (i < N_EXPERTS) cnt[i] = 0;
  for (; i < n; i += gridDim.x * blockDim.x) out[i] = 0.f;
}

// one wave per token: gate logits, top-2, softmax-of-2
__global__ void moe_gate_kernel(const float* __restrict__ x,
                                const float* __restrict__ wg,
                                const float* __restrict__ bg,
                                int* __restrict__ cnt,
                                int* __restrict__ tokE,
                                float* __restrict__ tokP) {
  int wave = blockIdx.x * (blockDim.x >> 5) + (threadIdx.x >> 5);
  int lane = threadIdx.x & 31;
  if (wave >= T_TOKENS) return;
  const float* xr = x + (size_t)wave * D_MODEL;
  float acc[N_EXPERTS];
#pragma unroll
  for (int e = 0; e < N_EXPERTS; ++e) acc[e] = 0.f;
  for (int d = lane; d < D_MODEL; d += 32) {
    float xv = xr[d];
    const float* wr = wg + (size_t)d * N_EXPERTS;
#pragma unroll
    for (int e = 0; e < N_EXPERTS; ++e) acc[e] += xv * wr[e];
  }
#pragma unroll
  for (int e = 0; e < N_EXPERTS; ++e) {
    float v = acc[e];
    for (int s = 16; s > 0; s >>= 1) v += __shfl_xor(v, s, 32);
    acc[e] = v + bg[e];
  }
  if (lane == 0) {
    int i0 = 0; float v0 = acc[0];
#pragma unroll
    for (int e = 1; e < N_EXPERTS; ++e)
      if (acc[e] > v0) { v0 = acc[e]; i0 = e; }
    int i1 = (i0 == 0) ? 1 : 0; float v1 = acc[i1];
#pragma unroll
    for (int e = 0; e < N_EXPERTS; ++e)
      if (e != i0 && acc[e] > v1) { v1 = acc[e]; i1 = e; }
    float p0 = 1.f / (1.f + __expf(v1 - v0));  // softmax over [v0, v1]
    tokE[2 * wave]     = i0;
    tokE[2 * wave + 1] = i1;
    tokP[2 * wave]     = p0;
    tokP[2 * wave + 1] = 1.f - p0;
    atomicAdd(&cnt[i0], 1);
    atomicAdd(&cnt[i1], 1);
  }
}

__global__ void moe_scan_kernel(const int* __restrict__ cnt,
                                int* __restrict__ offs,
                                int* __restrict__ fill) {
  if (threadIdx.x == 0 && blockIdx.x == 0) {
    int o = 0;
    for (int e = 0; e < N_EXPERTS; ++e) {
      offs[e] = o; fill[e] = o; o += cnt[e];
    }
    offs[N_EXPERTS] = o;
  }
}

__global__ void moe_scatter_kernel(const int* __restrict__ tokE,
                                   const float* __restrict__ tokP,
                                   int* __restrict__ fill,
                                   int* __restrict__ rows,
                                   float* __restrict__ gw) {
  int t = blockIdx.x * blockDim.x + threadIdx.x;
  if (t >= T_TOKENS) return;
#pragma unroll
  for (int s = 0; s < 2; ++s) {
    int e = tokE[2 * t + s];
    int pos = atomicAdd(&fill[e], 1);
    rows[pos] = t;
    gw[pos] = tokP[2 * t + s];
  }
}

__global__ void moe_cvt_kernel(const float* __restrict__ in,
                               unsigned short* __restrict__ out, int n) {
  for (int i = blockIdx.x * blockDim.x + threadIdx.x; i < n; i += gridDim.x * blockDim.x)
    out[i] = f2bf(in[i]);
}

// per-expert [R][C] fp32 -> [C][R] bf16 (K-contiguous for WMMA B tiles)
__global__ void moe_transpose_cvt_kernel(const float* __restrict__ in,
                                         unsigned short* __restrict__ out,
                                         int R, int C) {
  __shared__ unsigned short t[32][33];
  int e = blockIdx.z;
  int c0 = blockIdx.x * 32, r0 = blockIdx.y * 32;
  const float* src = in + (size_t)e * R * C;
  unsigned short* dst = out + (size_t)e * R * C;
  int tid = threadIdx.x;
#pragma unroll
  for (int j = 0; j < 4; ++j) {
    int idx = tid + j * 256; int r = idx >> 5, c = idx & 31;
    t[c][r] = f2bf(src[(size_t)(r0 + r) * C + (c0 + c)]);
  }
  __syncthreads();
#pragma unroll
  for (int j = 0; j < 4; ++j) {
    int idx = tid + j * 256; int r = idx >> 5, c = idx & 31;
    dst[(size_t)(c0 + r) * R + (r0 + c)] = t[r][c];
  }
}

// ---------- FFN1: h[slot, ff] = relu(x[tok] @ W1_e + b1_e), tile 64x128, K=1024 ----------
__global__ __launch_bounds__(256) void moe_ffn1_kernel(
    const unsigned short* __restrict__ xb,   // [T][1024] bf16
    const unsigned short* __restrict__ w1t,  // [E][4096][1024] bf16 (ff-major, K-contig)
    const float* __restrict__ b1,            // [E][4096]
    const int* __restrict__ offs,
    const int* __restrict__ rows,
    unsigned short* __restrict__ h) {        // [SLOTS][4096] bf16
  __shared__ unsigned short Al[2][64 * KST];
  __shared__ unsigned short Bl[2][128 * KST];
  __shared__ float b1s[128];
  __shared__ int tokS[64];

  int e = blockIdx.z;
  int off = offs[e], end = offs[e + 1];
  int m0 = blockIdx.y * 64;
  if (off + m0 >= end) return;
  int ff0 = blockIdx.x * 128;
  int tid = threadIdx.x;

  if (tid < 64) {
    int grow = off + m0 + tid;
    tokS[tid] = rows[grow < end ? grow : off];
  }
  if (tid < 128) b1s[tid] = b1[(size_t)e * D_FF + ff0 + tid];
  __syncthreads();

  int wave = tid >> 5, lane = tid & 31;
  int mStrip = wave & 3, nGroup = wave >> 2;   // 4 m-strips x 2 n-groups(64 each)
  v8f acc[4];
#pragma unroll
  for (int j = 0; j < 4; ++j) acc[j] = zero_v8f();

  // per-thread async-copy units (16B each): 1 for A, 2 for B
  int ar = tid >> 2, ap = tid & 3;
  const unsigned short* aSrc = xb + (size_t)tokS[ar] * D_MODEL + ap * 8;
  int r2 = (tid + 256) >> 2;
  const unsigned short* bSrc1 = w1t + ((size_t)e * D_FF + ff0 + ar) * D_MODEL + ap * 8;
  const unsigned short* bSrc2 = w1t + ((size_t)e * D_FF + ff0 + r2) * D_MODEL + ap * 8;
  const unsigned aOff  = ar * KST + ap * 8;
  const unsigned bOff1 = ar * KST + ap * 8;
  const unsigned bOff2 = r2 * KST + ap * 8;

  // issue loads for chunk at byte offset k (halves) into buffer nb (compile-time const)
  auto issue = [&](int nb, int k) {
    async_copy16(&Al[nb][aOff],  aSrc  + k);
    async_copy16(&Bl[nb][bOff1], bSrc1 + k);
    async_copy16(&Bl[nb][bOff2], bSrc2 + k);
  };
  // compute one K-chunk from buffer bsel (compile-time const)
  auto compute = [&](int bsel) {
    v16bf a  = load_frag_a(&Al[bsel][mStrip * 16 * KST], lane);
    v16bf b0 = load_frag_b(&Bl[bsel][(nGroup * 64 +  0) * KST], lane);
    v16bf b1f = load_frag_b(&Bl[bsel][(nGroup * 64 + 16) * KST], lane);
    v16bf b2f = load_frag_b(&Bl[bsel][(nGroup * 64 + 32) * KST], lane);
    v16bf b3f = load_frag_b(&Bl[bsel][(nGroup * 64 + 48) * KST], lane);
    acc[0] = __builtin_amdgcn_wmma_f32_16x16x32_bf16(false, a, false, b0,  (short)0, acc[0], false, false);
    acc[1] = __builtin_amdgcn_wmma_f32_16x16x32_bf16(false, a, false, b1f, (short)0, acc[1], false, false);
    acc[2] = __builtin_amdgcn_wmma_f32_16x16x32_bf16(false, a, false, b2f, (short)0, acc[2], false, false);
    acc[3] = __builtin_amdgcn_wmma_f32_16x16x32_bf16(false, a, false, b3f, (short)0, acc[3], false, false);
  };

  // software pipeline: NCH = 32 chunks (even); chunk i lives in buffer i&1
  issue(0, 0);
  int k = 32;
  for (int it = 0; it < (D_MODEL / 32) / 2 - 1; ++it) {
    issue(1, k);      WAIT_ASYNC(0x3); __syncthreads(); compute(0); __syncthreads();
    issue(0, k + 32); WAIT_ASYNC(0x3); __syncthreads(); compute(1); __syncthreads();
    k += 64;
  }
  issue(1, k); WAIT_ASYNC(0x3); __syncthreads(); compute(0); __syncthreads();
  WAIT_ASYNC(0x0); __syncthreads(); compute(1);

  int nl = lane & 15, half = lane >> 4;
#pragma unroll
  for (int j = 0; j < 4; ++j) {
    int ncol = nGroup * 64 + j * 16 + nl;
    float bias = b1s[ncol];
#pragma unroll
    for (int r = 0; r < 8; ++r) {
      int m = mStrip * 16 + r + half * 8;
      int grow = off + m0 + m;
      if (grow < end) {
        float v = acc[j][r] + bias;
        v = v > 0.f ? v : 0.f;
        h[(size_t)grow * D_FF + ff0 + ncol] = f2bf(v);
      }
    }
  }
}

// ---------- FFN2: out[tok,dm] += gate_w * (h[slot] @ W2_e + b2_e), tile 64x64, K=4096 ----------
__global__ __launch_bounds__(256) void moe_ffn2_kernel(
    const unsigned short* __restrict__ h,    // [SLOTS][4096] bf16
    const unsigned short* __restrict__ w2t,  // [E][1024][4096] bf16 (dm-major, K-contig)
    const float* __restrict__ b2,            // [E][1024]
    const int* __restrict__ offs,
    const int* __restrict__ rows,
    const float* __restrict__ gw,
    float* __restrict__ out) {               // [T][1024]
  __shared__ unsigned short Al[2][64 * KST];
  __shared__ unsigned short Bl[2][64 * KST];
  __shared__ float b2s[64];
  __shared__ float wS[64];
  __shared__ int tokS[64];
  __shared__ int srcS[64];

  int e = blockIdx.z;
  int off = offs[e], end = offs[e + 1];
  int m0 = blockIdx.y * 64;
  if (off + m0 >= end) return;
  int dm0 = blockIdx.x * 64;
  int tid = threadIdx.x;

  if (tid < 64) {
    int grow = off + m0 + tid;
    bool valid = grow < end;
    srcS[tid] = valid ? grow : off;
    tokS[tid] = valid ? rows[grow] : 0;
    wS[tid]   = valid ? gw[grow] : 0.f;
    b2s[tid]  = b2[(size_t)e * D_MODEL + dm0 + tid];
  }
  __syncthreads();

  int wave = tid >> 5, lane = tid & 31;
  int mStrip = wave & 3, nGroup = wave >> 2;   // 4 m-strips x 2 n-groups(32 each)
  v8f acc[2];
  acc[0] = zero_v8f(); acc[1] = zero_v8f();

  // per-thread async-copy units: 1 A + 1 B (64 rows x 4x16B)
  int r_ = tid >> 2, p_ = tid & 3;
  const unsigned short* aSrc = h + (size_t)srcS[r_] * D_FF + p_ * 8;
  const unsigned short* bSrc = w2t + ((size_t)e * D_MODEL + dm0 + r_) * D_FF + p_ * 8;
  const unsigned uOff = r_ * KST + p_ * 8;

  auto issue = [&](int nb, int k) {
    async_copy16(&Al[nb][uOff], aSrc + k);
    async_copy16(&Bl[nb][uOff], bSrc + k);
  };
  auto compute = [&](int bsel) {
    v16bf a  = load_frag_a(&Al[bsel][mStrip * 16 * KST], lane);
    v16bf b0 = load_frag_b(&Bl[bsel][(nGroup * 32 +  0) * KST], lane);
    v16bf b1f = load_frag_b(&Bl[bsel][(nGroup * 32 + 16) * KST], lane);
    acc[0] = __builtin_amdgcn_wmma_f32_16x16x32_bf16(false, a, false, b0,  (short)0, acc[0], false, false);
    acc[1] = __builtin_amdgcn_wmma_f32_16x16x32_bf16(false, a, false, b1f, (short)0, acc[1], false, false);
  };

  // software pipeline: NCH = 128 chunks (even); chunk i lives in buffer i&1
  issue(0, 0);
  int k = 32;
  for (int it = 0; it < (D_FF / 32) / 2 - 1; ++it) {
    issue(1, k);      WAIT_ASYNC(0x2); __syncthreads(); compute(0); __syncthreads();
    issue(0, k + 32); WAIT_ASYNC(0x2); __syncthreads(); compute(1); __syncthreads();
    k += 64;
  }
  issue(1, k); WAIT_ASYNC(0x2); __syncthreads(); compute(0); __syncthreads();
  WAIT_ASYNC(0x0); __syncthreads(); compute(1);

  int nl = lane & 15, half = lane >> 4;
#pragma unroll
  for (int j = 0; j < 2; ++j) {
    int ncol = nGroup * 32 + j * 16 + nl;
    float bias = b2s[ncol];
#pragma unroll
    for (int r = 0; r < 8; ++r) {
      int m = mStrip * 16 + r + half * 8;
      int grow = off + m0 + m;
      if (grow < end) {
        float v = (acc[j][r] + bias) * wS[m];
        atomicAdd(&out[(size_t)tokS[m] * D_MODEL + dm0 + ncol], v);
      }
    }
  }
}

// ---------- launch ----------

extern "C" void kernel_launch(void* const* d_in, const int* in_sizes, int n_in,
                              void* d_out, int out_size, void* d_ws, size_t ws_size,
                              hipStream_t stream) {
  const float* x  = (const float*)d_in[0];
  const float* wg = (const float*)d_in[1];
  const float* bg = (const float*)d_in[2];
  const float* w1 = (const float*)d_in[3];
  const float* b1 = (const float*)d_in[4];
  const float* w2 = (const float*)d_in[5];
  const float* b2 = (const float*)d_in[6];
  float* out = (float*)d_out;

  char* w = (char*)d_ws;
  int*   cnt   = (int*)(w + 0);                         // 8 ints
  int*   fill  = (int*)(w + 32);                        // 8 ints
  int*   offs  = (int*)(w + 64);                        // 9 ints
  int*   tokE  = (int*)(w + 128);                       // 4096 ints
  float* tokP  = (float*)(w + 128 + 16384);             // 4096 floats
  int*   rowsA = (int*)(w + 32896);                     // 4096 ints
  float* gwA   = (float*)(w + 49280);                   // 4096 floats
  size_t XB  = 65792;
  size_t W1T = XB  + (size_t)T_TOKENS * D_MODEL * 2;            // + 4 MB
  size_t W2T = W1T + (size_t)N_EXPERTS * D_MODEL * D_FF * 2;    // + 64 MB
  size_t HB  = W2T + (size_t)N_EXPERTS * D_FF * D_MODEL * 2;    // + 64 MB
  size_t NEED = HB + (size_t)SLOTS * D_FF * 2;                  // + 32 MB
  if (ws_size < NEED) return;
  unsigned short* xb  = (unsigned short*)(w + XB);
  unsigned short* w1t = (unsigned short*)(w + W1T);
  unsigned short* w2t = (unsigned short*)(w + W2T);
  unsigned short* hb  = (unsigned short*)(w + HB);

  const int nOut = T_TOKENS * D_MODEL;

  moe_init_kernel<<<dim3(8192), 256, 0, stream>>>(out, nOut, cnt);
  moe_gate_kernel<<<dim3(T_TOKENS / 8), 256, 0, stream>>>(x, wg, bg, cnt, tokE, tokP);
  moe_scan_kernel<<<dim3(1), 32, 0, stream>>>(cnt, offs, fill);
  moe_scatter_kernel<<<dim3(8), 256, 0, stream>>>(tokE, tokP, fill, rowsA, gwA);
  moe_cvt_kernel<<<dim3(4096), 256, 0, stream>>>(x, xb, nOut);
  // w1 [E][1024][4096] -> w1t [E][4096][1024]
  moe_transpose_cvt_kernel<<<dim3(D_FF / 32, D_MODEL / 32, N_EXPERTS), 256, 0, stream>>>(
      w1, w1t, D_MODEL, D_FF);
  // w2 [E][4096][1024] -> w2t [E][1024][4096]
  moe_transpose_cvt_kernel<<<dim3(D_MODEL / 32, D_FF / 32, N_EXPERTS), 256, 0, stream>>>(
      w2, w2t, D_FF, D_MODEL);
  moe_ffn1_kernel<<<dim3(D_FF / 128, SLOTS / 64, N_EXPERTS), 256, 0, stream>>>(
      xb, w1t, b1, offs, rowsA, hb);
  moe_ffn2_kernel<<<dim3(D_MODEL / 64, SLOTS / 64, N_EXPERTS), 256, 0, stream>>>(
      hb, w2t, b2, offs, rowsA, gwA, out);
}
